// ScaledDotProductAttention_23055384445500
// MI455X (gfx1250) — matrix operand order repaired and checked by reading, not compiled, not run on hardware
//
#include <hip/hip_runtime.h>

#define B_ 8
#define S_ 4096
#define E_ 256
#define H_ 128
#define KB 64
#define SCALE_QK 0.08838834764831845f  // 1/sqrt(128)

typedef __attribute__((ext_vector_type(16))) __bf16 v16bf;
typedef __attribute__((ext_vector_type(8)))  __bf16 v8bf;
typedef __attribute__((ext_vector_type(8)))  float  v8f;
typedef __attribute__((ext_vector_type(4)))  float  v4f;
typedef int v4i __attribute__((__vector_size__(4 * sizeof(int))));

#if __has_builtin(__builtin_amdgcn_global_load_async_to_lds_b128) && \
    __has_builtin(__builtin_amdgcn_s_wait_asynccnt)
#define HAVE_ASYNC 1
#else
#define HAVE_ASYNC 0
#endif

typedef __attribute__((address_space(1))) v4i* gv4ip;  // global (__device__)
typedef __attribute__((address_space(3))) v4i* lv4ip;  // LDS (__shared__)

// 16-byte global -> LDS copy: async DMA (ASYNCcnt) when available, else
// synchronous vector load/store (covered by the block barriers).
__device__ __forceinline__ void copy16_g2l(const __bf16* g, __bf16* l) {
#if HAVE_ASYNC
  __builtin_amdgcn_global_load_async_to_lds_b128((gv4ip)(void*)g,
                                                 (lv4ip)(void*)l, 0, 0);
#else
  *(v8bf*)l = *(const v8bf*)g;
#endif
}

__device__ __forceinline__ v8f wmma_bf16(v16bf a, v16bf b, v8f c) {
  return __builtin_amdgcn_wmma_f32_16x16x32_bf16(false, a, false, b, (short)0, c, false, false);
}

__device__ __forceinline__ v16bf pack16(v8bf lo, v8bf hi8) {
  v16bf t;
#pragma unroll
  for (int i = 0; i < 8; ++i) { t[i] = lo[i]; t[8 + i] = hi8[i]; }
  return t;
}

__device__ __forceinline__ v16bf cvt16(v4f a, v4f b, v4f c, v4f d) {
  v16bf t;
#pragma unroll
  for (int i = 0; i < 4; ++i) {
    t[i]      = (__bf16)a[i];
    t[4 + i]  = (__bf16)b[i];
    t[8 + i]  = (__bf16)c[i];
    t[12 + i] = (__bf16)d[i];
  }
  return t;
}

// ---------------------------------------------------------------------------
// Phase 1: Q = X Wq^T + bq, K = X Wk^T + bk, Vt = (X Wv^T + bv)^T   (bf16 out)
// Each wave: 16 rows of X (A-tile resident in regs), 3 x 8 N-tiles x 8 K-steps
// of v_wmma_f32_16x16x32_bf16. Q,K stored [B*S,H]; V stored transposed [B,H,S].
// ---------------------------------------------------------------------------
__global__ __launch_bounds__(128) void proj_kernel(
    const float* __restrict__ X,
    const float* __restrict__ Wq, const float* __restrict__ bq,
    const float* __restrict__ Wk, const float* __restrict__ bk,
    const float* __restrict__ Wv, const float* __restrict__ bv,
    __bf16* __restrict__ Qo, __bf16* __restrict__ Ko, __bf16* __restrict__ Vt)
{
  const int lane = threadIdx.x & 31;
  const int wave = threadIdx.x >> 5;
  const int m    = lane & 15;
  const int hi   = lane >> 4;
  const int r0   = (blockIdx.x * 4 + wave) * 16;   // 16 rows of flattened [B*S]
  const int bidx = r0 >> 12;
  const int s0   = r0 & (S_ - 1);

  v16bf a[8];
  const float* xrow = X + (size_t)(r0 + m) * E_;
#pragma unroll
  for (int ks = 0; ks < 8; ++ks) {
    const float* p = xrow + ks * 32 + hi * 8;
    v4f c0 = *(const v4f*)(p);
    v4f c1 = *(const v4f*)(p + 4);
    v4f c2 = *(const v4f*)(p + 16);
    v4f c3 = *(const v4f*)(p + 20);
    a[ks] = cvt16(c0, c1, c2, c3);
  }

  const float* Ws[3]   = {Wq, Wk, Wv};
  const float* bias[3] = {bq, bk, bv};

#pragma unroll
  for (int o = 0; o < 3; ++o) {
#pragma unroll
    for (int nt = 0; nt < 8; ++nt) {
      const int n = nt * 16 + m;
      const float b0 = bias[o][n];
      v8f acc;
#pragma unroll
      for (int j = 0; j < 8; ++j) acc[j] = b0;

      const float* wrow = Ws[o] + (size_t)n * E_;
#pragma unroll
      for (int ks = 0; ks < 8; ++ks) {
        const float* p = wrow + ks * 32 + hi * 8;
        v4f c0 = *(const v4f*)(p);
        v4f c1 = *(const v4f*)(p + 4);
        v4f c2 = *(const v4f*)(p + 16);
        v4f c3 = *(const v4f*)(p + 20);
        acc = wmma_bf16(a[ks], cvt16(c0, c1, c2, c3), acc);
      }

      if (o == 0) {
#pragma unroll
        for (int j = 0; j < 8; ++j)
          Qo[(size_t)(r0 + j + hi * 8) * H_ + n] = (__bf16)acc[j];
      } else if (o == 1) {
#pragma unroll
        for (int j = 0; j < 8; ++j)
          Ko[(size_t)(r0 + j + hi * 8) * H_ + n] = (__bf16)acc[j];
      } else {
#pragma unroll
        for (int j = 0; j < 8; ++j)
          Vt[(size_t)bidx * H_ * S_ + (size_t)n * S_ + (s0 + j + hi * 8)] =
              (__bf16)acc[j];
      }
    }
  }
}

// ---------------------------------------------------------------------------
// Phase 2: flash attention with double-buffered async K/V staging in LDS.
// Block = 4 waves x 16 query rows, batch-resident. Per 64-key block:
//   async DMA next K (64x128) + V^T (128x64) tiles -> LDS (ASYNCcnt),
//   16 WMMAs for S = Q K^T, online softmax (16-lane butterflies + v_exp_f32),
//   P transposed through per-wave LDS tile, 16 WMMAs for O += P V.
// ---------------------------------------------------------------------------
__global__ __launch_bounds__(128) void attn_kernel(
    const __bf16* __restrict__ Q, const __bf16* __restrict__ K,
    const __bf16* __restrict__ Vt, float* __restrict__ O)
{
  __shared__ __bf16 Ktile[2][KB * H_];   // [key][h]   16 KB each
  __shared__ __bf16 Vtile[2][H_ * KB];   // [h][key]   16 KB each
  __shared__ __bf16 plds[4][16 * KB];    // per-wave P staging, 2 KB each

  const int tid  = threadIdx.x;
  const int lane = tid & 31;
  const int wave = tid >> 5;
  const int m    = lane & 15;
  const int hi   = lane >> 4;
  const int blocksPerBatch = S_ / 64;
  const int b  = blockIdx.x / blocksPerBatch;
  const int qt = blockIdx.x % blocksPerBatch;
  const int qbase = qt * 64 + wave * 16;

  const __bf16* Kb = K + (size_t)b * S_ * H_;
  const __bf16* Vb = Vt + (size_t)b * H_ * S_;
  __bf16* pw = &plds[wave][0];

  // Cooperative tile fills: 16 x b128 per wave per (K,V) pair.
  auto fillK = [&](int buf, int kb) {
#pragma unroll
    for (int i = 0; i < 8; ++i) {               // 64*128 elems, 16B per thread
      const int off = (i * 128 + tid) * 8;
      copy16_g2l(Kb + (size_t)kb * H_ + off, &Ktile[buf][off]);
    }
  };
  auto fillV = [&](int buf, int kb) {
#pragma unroll
    for (int i = 0; i < 8; ++i) {               // thread t owns row h=t
      copy16_g2l(Vb + (size_t)tid * S_ + kb + i * 8,
                 &Vtile[buf][tid * KB + i * 8]);
    }
  };

  // Q A-operands for h = 0..127 (4 K-steps of 32), resident all kernel.
  v16bf qa[4];
  const __bf16* qrow = Q + (size_t)(b * S_ + qbase + m) * H_;
#pragma unroll
  for (int ks = 0; ks < 4; ++ks) {
    const __bf16* p = qrow + ks * 32 + hi * 8;
    qa[ks] = pack16(*(const v8bf*)(p), *(const v8bf*)(p + 16));
  }

  float mrow[8], lrow[8];
#pragma unroll
  for (int j = 0; j < 8; ++j) { mrow[j] = -3.0e38f; lrow[j] = 0.0f; }
  v8f oacc[8];
#pragma unroll
  for (int ht = 0; ht < 8; ++ht)
#pragma unroll
    for (int j = 0; j < 8; ++j) oacc[ht][j] = 0.0f;

  // Prologue: stage first tile.
  fillK(0, 0);
  fillV(0, 0);

  for (int kb = 0; kb < S_; kb += KB) {
    const int cur = (kb >> 6) & 1;
    const int nxt = cur ^ 1;

    // Kick off DMA for the next tile, then wait for the current one.
    if (kb + KB < S_) {
      fillK(nxt, kb + KB);
      fillV(nxt, kb + KB);
#if HAVE_ASYNC
      __builtin_amdgcn_s_wait_asynccnt(16);   // in-order: current tile done
#endif
    } else {
#if HAVE_ASYNC
      __builtin_amdgcn_s_wait_asynccnt(0);
#endif
    }
    __syncthreads();   // all waves' fills of `cur` visible

    // ---- scores: 4 N-tiles x 4 K-steps of WMMA from LDS
    v8f sacc[4];
#pragma unroll
    for (int nt = 0; nt < 4; ++nt) {
      v8f acc;
#pragma unroll
      for (int j = 0; j < 8; ++j) acc[j] = 0.0f;
      const __bf16* krow = &Ktile[cur][(nt * 16 + m) * H_ + hi * 8];
#pragma unroll
      for (int ks = 0; ks < 4; ++ks) {
        const __bf16* p = krow + ks * 32;
        acc = wmma_bf16(qa[ks], pack16(*(const v8bf*)(p), *(const v8bf*)(p + 16)), acc);
      }
      sacc[nt] = acc;
    }

#pragma unroll
    for (int nt = 0; nt < 4; ++nt) sacc[nt] *= SCALE_QK;

    // ---- online softmax (per query row = (j, hi))
#pragma unroll
    for (int j = 0; j < 8; ++j) {
      float t0 = fmaxf(fmaxf(sacc[0][j], sacc[1][j]), fmaxf(sacc[2][j], sacc[3][j]));
#pragma unroll
      for (int w = 1; w < 16; w <<= 1) t0 = fmaxf(t0, __shfl_xor(t0, w, 16));
      const float mn    = fmaxf(mrow[j], t0);
      const float alpha = __expf(mrow[j] - mn);
      mrow[j] = mn;
      float rs = 0.0f;
#pragma unroll
      for (int nt = 0; nt < 4; ++nt) {
        const float pp = __expf(sacc[nt][j] - mn);
        sacc[nt][j] = pp;
        rs += pp;
      }
#pragma unroll
      for (int w = 1; w < 16; w <<= 1) rs += __shfl_xor(rs, w, 16);
      lrow[j] = lrow[j] * alpha + rs;
#pragma unroll
      for (int ht = 0; ht < 8; ++ht) oacc[ht][j] *= alpha;
    }

    // ---- P (C-layout) -> per-wave LDS tile -> A-operand layout
#pragma unroll
    for (int nt = 0; nt < 4; ++nt)
#pragma unroll
      for (int j = 0; j < 8; ++j)
        pw[(j + hi * 8) * KB + nt * 16 + m] = (__bf16)sacc[nt][j];

    v16bf pa[2];
#pragma unroll
    for (int a2 = 0; a2 < 2; ++a2) {
      const __bf16* p = pw + m * KB + a2 * 32 + hi * 8;
      pa[a2] = pack16(*(const v8bf*)(p), *(const v8bf*)(p + 16));
    }

    // ---- O += P V : 8 h-tiles x 2 K-steps from LDS
#pragma unroll
    for (int ht = 0; ht < 8; ++ht) {
      const __bf16* vcol = &Vtile[cur][(ht * 16 + m) * KB + hi * 8];
#pragma unroll
      for (int a2 = 0; a2 < 2; ++a2) {
        const __bf16* p = vcol + a2 * 32;
        oacc[ht] = wmma_bf16(pa[a2],
                             pack16(*(const v8bf*)(p), *(const v8bf*)(p + 16)),
                             oacc[ht]);
      }
    }

    __syncthreads();   // all reads of `cur` done before next fill overwrites it
  }

  // ---- finalize: O /= l, store f32 row-major [B,S,H]
  float* orow = O + (size_t)(b * S_ + qbase) * H_;
#pragma unroll
  for (int j = 0; j < 8; ++j) {
    const float inv = 1.0f / lrow[j];
#pragma unroll
    for (int ht = 0; ht < 8; ++ht)
      orow[(size_t)(j + hi * 8) * H_ + ht * 16 + m] = oacc[ht][j] * inv;
  }
}

// ---------------------------------------------------------------------------
extern "C" void kernel_launch(void* const* d_in, const int* in_sizes, int n_in,
                              void* d_out, int out_size, void* d_ws, size_t ws_size,
                              hipStream_t stream) {
  (void)in_sizes; (void)n_in; (void)out_size; (void)ws_size;
  const float* X  = (const float*)d_in[0];
  const float* Wk = (const float*)d_in[1];
  const float* bk = (const float*)d_in[2];
  const float* Wq = (const float*)d_in[3];
  const float* bq = (const float*)d_in[4];
  const float* Wv = (const float*)d_in[5];
  const float* bv = (const float*)d_in[6];
  float* out = (float*)d_out;

  __bf16* Qb = (__bf16*)d_ws;
  __bf16* Kb = Qb + (size_t)B_ * S_ * H_;
  __bf16* Vt = Kb + (size_t)B_ * S_ * H_;

  proj_kernel<<<512, 128, 0, stream>>>(X, Wq, bq, Wk, bk, Wv, bv, Qb, Kb, Vt);
  attn_kernel<<<512, 128, 0, stream>>>(Qb, Kb, Vt, out);
}
